// FunnelAttention_67817533604276
// MI455X (gfx1250) — compile-verified, matrix operand order checked by reading, not run on hardware
//
#include <hip/hip_runtime.h>

// ---------------------------------------------------------------------------
// Types / helpers
// ---------------------------------------------------------------------------
typedef unsigned short u16;
typedef __attribute__((ext_vector_type(16))) __bf16 v16bf;
typedef __attribute__((ext_vector_type(8)))  float  v8f;
typedef __attribute__((ext_vector_type(4)))  unsigned uv4;
typedef __attribute__((ext_vector_type(8)))  int      iv8;
typedef __attribute__((ext_vector_type(4)))  int      iv4;

struct alignas(16) U4 { unsigned x, y, z, w; };
struct FragBits { U4 lo, hi; };

#if defined(__has_builtin)
#if __has_builtin(__builtin_amdgcn_tensor_load_to_lds) && \
    __has_builtin(__builtin_amdgcn_s_wait_tensorcnt)
#define USE_TDM 1
#endif
#endif
#ifndef USE_TDM
#define USE_TDM 0
#endif

__device__ __forceinline__ float bf2f(u16 h) {
    unsigned u = ((unsigned)h) << 16;
    return __builtin_bit_cast(float, u);
}
__device__ __forceinline__ u16 f2bf(float f) {
    unsigned u = __builtin_bit_cast(unsigned, f);
    u += 0x7FFFu + ((u >> 16) & 1u);   // round-to-nearest-even
    return (u16)(u >> 16);
}

// Fragment layout (16x16x32 bf16 operand, wave32):
//   lane m = lane%16 selects row; per lane K-chunks: [half*8 .. +7] and [16+half*8 .. +7]
__device__ __forceinline__ v16bf make_frag(const u16* p) {
    FragBits f;
    f.lo = *(const U4*)p;
    f.hi = *(const U4*)(p + 16);
    return __builtin_bit_cast(v16bf, f);
}
__device__ __forceinline__ v16bf zero_frag() {
    FragBits f = {};
    return __builtin_bit_cast(v16bf, f);
}
__device__ __forceinline__ v8f wmma_bf16(v16bf a, v16bf b, v8f c) {
    return __builtin_amdgcn_wmma_f32_16x16x32_bf16(false, a, false, b, (short)0, c,
                                                   false, false);
}
__device__ __forceinline__ v8f zero8() {
    v8f a;
    #pragma unroll
    for (int i = 0; i < 8; ++i) a[i] = 0.f;
    return a;
}

#if USE_TDM
// TDM: DMA a 2D tile of 128 rows x 64 bf16 (row stride ld_elems) from global
// into LDS at lds_off, padding every 32 DWORDs (=1 row) by 4 DWORDs so the
// effective LDS row stride is 72 elems (144B) -- matches the WMMA frag layout.
// Rows >= rows_avail read as zero (hardware OOB fill). D# per CDNA5 ISA.
__device__ __forceinline__ void tdm_load_tile(const u16* gptr, unsigned lds_off,
                                              int ld_elems, int rows_avail) {
    unsigned long long ga = (unsigned long long)(size_t)gptr;
    uv4 g0;
    g0[0] = 1u;                                   // count=1, user descriptor
    g0[1] = lds_off;                              // lds_addr (bytes)
    g0[2] = (unsigned)ga;                         // global_addr[31:0]
    g0[3] = (unsigned)((ga >> 32) & 0x1FFFFFFu)   // global_addr[56:32]
          | (2u << 30);                           // type=2 ("image")
    iv8 g1;
    g1[0] = (1 << 16)      // data_size = 2 bytes
          | (1 << 20)      // pad_enable
          | (4 << 22)      // pad_interval: 32 DWORDs
          | (3 << 25);     // pad_amount: 4 DWORDs
    g1[1] = (int)(64u << 16);                               // tensor_dim0 = 64
    g1[2] = (int)(((unsigned)rows_avail & 0xFFFFu) << 16);  // tensor_dim1 lo
    g1[3] = (int)((((unsigned)rows_avail >> 16) & 0xFFFFu)  // tensor_dim1 hi
          | (64u << 16));                                   // tile_dim0 = 64
    g1[4] = 128;                                            // tile_dim1 = 128
    g1[5] = ld_elems;                                       // dim0_stride lo
    g1[6] = 0;
    g1[7] = 0;
    iv4 z4 = {0, 0, 0, 0};
#if __clang_major__ >= 23
    iv8 z8 = {0, 0, 0, 0, 0, 0, 0, 0};
    __builtin_amdgcn_tensor_load_to_lds(g0, g1, z4, z4, z8, 0);
#else
    __builtin_amdgcn_tensor_load_to_lds(g0, g1, z4, z4, 0);
#endif
}
#endif

// ---------------------------------------------------------------------------
// f32 -> bf16 bulk convert (same layout)
// ---------------------------------------------------------------------------
__global__ __launch_bounds__(256) void cvt_kernel(const float* __restrict__ in,
                                                  u16* __restrict__ out, long n) {
    long i = (long)blockIdx.x * 256 + threadIdx.x;
    long stride = (long)gridDim.x * 256;
    for (; i < n; i += stride) out[i] = f2bf(in[i]);
}

// f32 -> bf16 with transpose: in (batch, rows, cols) -> out (batch, cols, rows)
__global__ __launch_bounds__(256)
void cvtT_kernel(const float* __restrict__ in, u16* __restrict__ out,
                 long rows, long cols, long total) {
    long i = (long)blockIdx.x * 256 + threadIdx.x;
    long stride = (long)gridDim.x * 256;
    long rc = rows * cols;
    for (; i < total; i += stride) {
        long b = i / rc, rem = i - b * rc;
        long r = rem / cols, c = rem - r * cols;
        out[b * rc + c * rows + r] = f2bf(in[i]);
    }
}

// ---------------------------------------------------------------------------
// bf16 WMMA GEMM: C[M,N] = A[M,K] @ B[K,N] (+bias[N]), B given TRANSPOSED:
// BT is (N,K) row-major. Block 256 thr = 8 waves, tile 128x128, K-step 64,
// double-buffered LDS. Tile staging: TDM DMA (wave 0 issues, TENSORcnt) when
// available, else VMEM loads + LDS stores with GL2 prefetch.
// Wave (wr=w>>1 in 0..3, wc=w&1) computes 32x64 = 2x4 subtiles of 16x16.
// ---------------------------------------------------------------------------
template <bool OUT_F32>
__global__ __launch_bounds__(256)
void gemm_bf16_kernel(const u16* __restrict__ A, long batchA,
                      const u16* __restrict__ BT, long batchB,
                      void* __restrict__ Cp, long batchC,
                      const float* __restrict__ bias,
                      int M, int N, int K, int lda, int ldbt, int ldc) {
    __shared__ u16 sA[2][128 * 72];   // [m][k0..63], stride 72 (144B, 16B-aligned)
    __shared__ u16 sB[2][128 * 72];   // [n][k0..63]

    const long zb = blockIdx.z;
    A  += zb * batchA;
    BT += zb * batchB;
    const int m0 = blockIdx.y * 128, n0 = blockIdx.x * 128;
    const int t = threadIdx.x, w = t >> 5, lane = t & 31;
    const int hlf = (lane >> 4) & 1, l16 = lane & 15;
    const int wr = w >> 1, wc = w & 1;

    v8f acc[2][4];
    #pragma unroll
    for (int i = 0; i < 2; ++i)
        #pragma unroll
        for (int j = 0; j < 4; ++j) acc[i][j] = zero8();

    const int nsteps = K >> 6;   // K multiple of 64 for all uses

#if USE_TDM
    const unsigned sAoff0 = (unsigned)(size_t)&sA[0][0];
    const unsigned sBoff0 = (unsigned)(size_t)&sB[0][0];
    const unsigned bufBytes = 128 * 72 * 2;
    if (w == 0) {
        tdm_load_tile(A + (long)m0 * lda, sAoff0, lda, M - m0);
        tdm_load_tile(BT + (long)n0 * ldbt, sBoff0, ldbt, N - n0);
    }
    for (int s = 0; s < nsteps; ++s) {
        const int cur = s & 1, nxt = cur ^ 1;
        if (w == 0) __builtin_amdgcn_s_wait_tensorcnt(0);  // tile s landed
        __syncthreads();            // publish tile s; buf[nxt] reads all done
        if (w == 0 && s + 1 < nsteps) {
            tdm_load_tile(A + (long)m0 * lda + (s + 1) * 64,
                          sAoff0 + nxt * bufBytes, lda, M - m0);
            tdm_load_tile(BT + (long)n0 * ldbt + (s + 1) * 64,
                          sBoff0 + nxt * bufBytes, ldbt, N - n0);
        }
        #pragma unroll
        for (int kc = 0; kc < 2; ++kc) {
            v16bf af[2], bfr[4];
            #pragma unroll
            for (int i = 0; i < 2; ++i)
                af[i] = make_frag(&sA[cur][(wr * 32 + i * 16 + l16) * 72 + kc * 32 + hlf * 8]);
            #pragma unroll
            for (int j = 0; j < 4; ++j)
                bfr[j] = make_frag(&sB[cur][(wc * 64 + j * 16 + l16) * 72 + kc * 32 + hlf * 8]);
            #pragma unroll
            for (int i = 0; i < 2; ++i)
                #pragma unroll
                for (int j = 0; j < 4; ++j)
                    acc[i][j] = wmma_bf16(af[i], bfr[j], acc[i][j]);
        }
    }
#else
    const int lr = t >> 3;          // 0..31 (row group)
    const int lc = (t & 7) * 8;     // 0..56 (16B column group)
    auto loadA = [&](int k0, U4* regs) {
        #pragma unroll
        for (int it = 0; it < 4; ++it) {
            int gm = m0 + lr + it * 32;
            U4 z = {0, 0, 0, 0};
            regs[it] = (gm < M) ? *(const U4*)(A + (long)gm * lda + k0 + lc) : z;
        }
    };
    auto loadB = [&](int k0, U4* regs) {
        #pragma unroll
        for (int it = 0; it < 4; ++it) {
            int gn = n0 + lr + it * 32;
            U4 z = {0, 0, 0, 0};
            regs[it] = (gn < N) ? *(const U4*)(BT + (long)gn * ldbt + k0 + lc) : z;
        }
    };
    auto storeTiles = [&](int buf, const U4* ra, const U4* rb) {
        #pragma unroll
        for (int it = 0; it < 4; ++it) {
            *(U4*)&sA[buf][(lr + it * 32) * 72 + lc] = ra[it];
            *(U4*)&sB[buf][(lr + it * 32) * 72 + lc] = rb[it];
        }
    };
    U4 ra[4], rb[4];
    loadA(0, ra);
    loadB(0, rb);
    storeTiles(0, ra, rb);
    for (int s = 0; s < nsteps; ++s) {
        const int cur = s & 1, nxt = cur ^ 1;
        if (s + 1 < nsteps) {
            loadA((s + 1) * 64, ra);
            loadB((s + 1) * 64, rb);
        }
        if (s + 2 < nsteps) {
            int gm = m0 + lr, gn = n0 + lr;
            if (gm < M) __builtin_prefetch(A + (long)gm * lda + (s + 2) * 64 + lc, 0, 1);
            if (gn < N) __builtin_prefetch(BT + (long)gn * ldbt + (s + 2) * 64 + lc, 0, 1);
        }
        __syncthreads();
        #pragma unroll
        for (int kc = 0; kc < 2; ++kc) {
            v16bf af[2], bfr[4];
            #pragma unroll
            for (int i = 0; i < 2; ++i)
                af[i] = make_frag(&sA[cur][(wr * 32 + i * 16 + l16) * 72 + kc * 32 + hlf * 8]);
            #pragma unroll
            for (int j = 0; j < 4; ++j)
                bfr[j] = make_frag(&sB[cur][(wc * 64 + j * 16 + l16) * 72 + kc * 32 + hlf * 8]);
            #pragma unroll
            for (int i = 0; i < 2; ++i)
                #pragma unroll
                for (int j = 0; j < 4; ++j)
                    acc[i][j] = wmma_bf16(af[i], bfr[j], acc[i][j]);
        }
        if (s + 1 < nsteps) storeTiles(nxt, ra, rb);
    }
#endif

    // Epilogue (C layout: col = l16, rows = v + 8*hlf)
    #pragma unroll
    for (int i = 0; i < 2; ++i)
        #pragma unroll
        for (int j = 0; j < 4; ++j) {
            int col = n0 + wc * 64 + j * 16 + l16;
            if (col >= N) continue;
            float bv = bias ? bias[col] : 0.f;
            #pragma unroll
            for (int v = 0; v < 8; ++v) {
                int row = m0 + wr * 32 + i * 16 + v + 8 * hlf;
                if (row >= M) continue;
                float val = acc[i][j][v] + bv;
                if (OUT_F32)
                    ((float*)Cp)[zb * batchC + (long)row * ldc + col] = val;
                else
                    ((u16*)Cp)[zb * batchC + (long)row * ldc + col] = f2bf(val);
            }
        }
}

// ---------------------------------------------------------------------------
// Flash attention with banded relative-position term.
// pos'[i,j] = Qr[i] . r[L + j - i]  (derived from the reference's shift trick)
// Block: (b, n, 128-query tile), 8 waves; wave w owns query rows w*16..+15.
// qh/kh/vh layout (b,l,n,h); rmat (n, 2L, H); obf (b,l,n,h) bf16.
// ---------------------------------------------------------------------------
__global__ __launch_bounds__(256)
void attn_kernel(const u16* __restrict__ qh, const u16* __restrict__ kh,
                 const u16* __restrict__ vh, const u16* __restrict__ rmat,
                 const unsigned char* __restrict__ pad,
                 const float* __restrict__ rwb, const float* __restrict__ rrb,
                 u16* __restrict__ obf) {
    const int L = 1024, NH = 16, H = 64;
    __shared__ u16  sVT[64 * 136];     // V transposed: [h][key]
    __shared__ u16  sP[128 * 136];     // P bf16 (wave-private 16-row slabs)
    __shared__ float sPos[128 * 129];  // banded pos, one 128-col half at a time

    const int i0 = blockIdx.x * 128;
    const int n = blockIdx.y, b = blockIdx.z;
    const int t = threadIdx.x, w = t >> 5, lane = t & 31;
    const int hlf = (lane >> 4) & 1, l16 = lane & 15;
    const float scale = 0.125f;  // 1/sqrt(64)

    // Q fragments with r_w_bias / r_r_bias folded in (built once, registers)
    v16bf qw[2], qr[2];
    {
        const u16* qrow = qh + (((long)b * L + i0 + w * 16 + l16) * NH + n) * H;
        #pragma unroll
        for (int ch = 0; ch < 2; ++ch) {
            alignas(16) u16 tw[16], tr[16];
            const u16* p = qrow + ch * 32 + hlf * 8;
            #pragma unroll
            for (int e = 0; e < 8; ++e) {
                int h = ch * 32 + hlf * 8 + e;
                float f = bf2f(p[e]);
                tw[e] = f2bf(f + rwb[n * H + h]);
                tr[e] = f2bf(f + rrb[n * H + h]);
            }
            #pragma unroll
            for (int e = 0; e < 8; ++e) {
                int h = ch * 32 + 16 + hlf * 8 + e;
                float f = bf2f(p[16 + e]);
                tw[8 + e] = f2bf(f + rwb[n * H + h]);
                tr[8 + e] = f2bf(f + rrb[n * H + h]);
            }
            FragBits fw, fr;
            fw.lo = *(U4*)&tw[0]; fw.hi = *(U4*)&tw[8];
            fr.lo = *(U4*)&tr[0]; fr.hi = *(U4*)&tr[8];
            qw[ch] = __builtin_bit_cast(v16bf, fw);
            qr[ch] = __builtin_bit_cast(v16bf, fr);
        }
    }

    v8f oacc[4];
    #pragma unroll
    for (int ot = 0; ot < 4; ++ot) oacc[ot] = zero8();
    float mrow[8], lrow[8];
    #pragma unroll
    for (int v = 0; v < 8; ++v) { mrow[v] = -1e30f; lrow[v] = 0.f; }

    for (int j0 = 0; j0 < L; j0 += 128) {
        __syncthreads();  // protect sVT against next overwrite
        // Cooperative V transpose load: 128 keys x 64 h -> sVT[h][key]
        {
            int col = (t & 7) * 8;
            #pragma unroll
            for (int it = 0; it < 4; ++it) {
                int r = (t >> 3) + it * 32;
                const u16* src = vh + (((long)b * L + j0 + r) * NH + n) * H + col;
                alignas(16) u16 e[8];
                *(U4*)e = *(const U4*)src;
                #pragma unroll
                for (int j = 0; j < 8; ++j) sVT[(col + j) * 136 + r] = e[j];
            }
        }
        __syncthreads();

        // S = Qw @ K^T  (K fragments straight from global rows)
        float sreg[8][8];
        #pragma unroll
        for (int tt = 0; tt < 8; ++tt) {
            const u16* krow = kh + (((long)b * L + j0 + tt * 16 + l16) * NH + n) * H;
            v8f acc = zero8();
            acc = wmma_bf16(qw[0], make_frag(krow + hlf * 8), acc);
            acc = wmma_bf16(qw[1], make_frag(krow + 32 + hlf * 8), acc);
            #pragma unroll
            for (int v = 0; v < 8; ++v) sreg[tt][v] = acc[v];
        }

        // Banded pos term in two 128-column passes through sPos (wave-private rows)
        const int c0 = L + j0 - i0 - 127;
        #pragma unroll
        for (int ph = 0; ph < 2; ++ph) {
            #pragma unroll
            for (int ct = 0; ct < 8; ++ct) {
                int cbase = c0 + ph * 128 + ct * 16 + l16;
                v16bf b0, b1;
                if ((unsigned)cbase < 2048u) {
                    const u16* rrow = rmat + ((long)n * 2048 + cbase) * H;
                    b0 = make_frag(rrow + hlf * 8);
                    b1 = make_frag(rrow + 32 + hlf * 8);
                } else {
                    b0 = zero_frag();
                    b1 = zero_frag();
                }
                v8f acc = zero8();
                acc = wmma_bf16(qr[0], b0, acc);
                acc = wmma_bf16(qr[1], b1, acc);
                #pragma unroll
                for (int v = 0; v < 8; ++v)
                    sPos[(w * 16 + v + 8 * hlf) * 129 + ct * 16 + l16] = acc[v];
            }
            // gather: score(di, j) += pos[di][ (j-j0) - di + 127 ]
            #pragma unroll
            for (int tt = 0; tt < 8; ++tt)
                #pragma unroll
                for (int v = 0; v < 8; ++v) {
                    int di = w * 16 + v + 8 * hlf;
                    int idx = tt * 16 + l16 - di + 127;
                    if ((idx >> 7) == ph) sreg[tt][v] += sPos[di * 129 + (idx & 127)];
                }
        }

        // scale + padding mask + online softmax
        float padv[8];
        #pragma unroll
        for (int tt = 0; tt < 8; ++tt)
            padv[tt] = pad[(long)b * L + j0 + tt * 16 + l16] ? 1e6f : 0.f;
        float mnew[8];
        #pragma unroll
        for (int v = 0; v < 8; ++v) mnew[v] = mrow[v];
        #pragma unroll
        for (int tt = 0; tt < 8; ++tt)
            #pragma unroll
            for (int v = 0; v < 8; ++v) {
                float s = sreg[tt][v] * scale - padv[tt];
                sreg[tt][v] = s;
                mnew[v] = fmaxf(mnew[v], s);
            }
        #pragma unroll
        for (int off = 1; off < 16; off <<= 1)
            #pragma unroll
            for (int v = 0; v < 8; ++v)
                mnew[v] = fmaxf(mnew[v], __shfl_xor(mnew[v], off, 32));
        float alpha[8], psum[8];
        #pragma unroll
        for (int v = 0; v < 8; ++v) {
            alpha[v] = __expf(mrow[v] - mnew[v]);
            psum[v] = 0.f;
            mrow[v] = mnew[v];
        }
        #pragma unroll
        for (int tt = 0; tt < 8; ++tt)
            #pragma unroll
            for (int v = 0; v < 8; ++v) {
                int di = w * 16 + v + 8 * hlf;
                float p = __expf(sreg[tt][v] - mnew[v]);
                psum[v] += p;
                sP[di * 136 + tt * 16 + l16] = f2bf(p);
            }
        #pragma unroll
        for (int off = 1; off < 16; off <<= 1)
            #pragma unroll
            for (int v = 0; v < 8; ++v) psum[v] += __shfl_xor(psum[v], off, 32);
        #pragma unroll
        for (int v = 0; v < 8; ++v) lrow[v] = lrow[v] * alpha[v] + psum[v];
        #pragma unroll
        for (int ot = 0; ot < 4; ++ot)
            #pragma unroll
            for (int v = 0; v < 8; ++v) oacc[ot][v] *= alpha[v];

        // O += P @ V   (P re-fragmented through wave-private sP rows)
        #pragma unroll
        for (int kc = 0; kc < 4; ++kc) {
            v16bf a = make_frag(&sP[(w * 16 + l16) * 136 + kc * 32 + hlf * 8]);
            #pragma unroll
            for (int ot = 0; ot < 4; ++ot) {
                v16bf bb = make_frag(&sVT[(ot * 16 + l16) * 136 + kc * 32 + hlf * 8]);
                oacc[ot] = wmma_bf16(a, bb, oacc[ot]);
            }
        }
    }

    // Finalize: O /= l, store bf16 (b,l,n,h)
    #pragma unroll
    for (int ot = 0; ot < 4; ++ot)
        #pragma unroll
        for (int v = 0; v < 8; ++v) {
            int di = w * 16 + v + 8 * hlf;
            float val = oacc[ot][v] / lrow[v];
            obf[(((long)b * L + i0 + di) * NH + n) * H + ot * 16 + l16] = f2bf(val);
        }
}

// ---------------------------------------------------------------------------
// Residual + LayerNorm over rows of 1024
// ---------------------------------------------------------------------------
__global__ __launch_bounds__(256)
void ln_kernel(const float* __restrict__ x0, const float* __restrict__ att,
               const float* __restrict__ g, const float* __restrict__ bb,
               float* __restrict__ out) {
    long row = blockIdx.x;
    const float* a = x0 + row * 1024;
    const float* c = att + row * 1024;
    float* o = out + row * 1024;
    int t = threadIdx.x;
    float x[4], s = 0.f, s2 = 0.f;
    #pragma unroll
    for (int i = 0; i < 4; ++i) {
        int idx = i * 256 + t;
        float v = a[idx] + c[idx];
        x[i] = v; s += v; s2 += v * v;
    }
    #pragma unroll
    for (int off = 16; off; off >>= 1) {
        s += __shfl_xor(s, off, 32);
        s2 += __shfl_xor(s2, off, 32);
    }
    __shared__ float rs[8], rs2[8];
    int w = t >> 5, lane = t & 31;
    if (lane == 0) { rs[w] = s; rs2[w] = s2; }
    __syncthreads();
    float ts = 0.f, ts2 = 0.f;
    #pragma unroll
    for (int i = 0; i < 8; ++i) { ts += rs[i]; ts2 += rs2[i]; }
    float mu = ts * (1.f / 1024.f);
    float var = ts2 * (1.f / 1024.f) - mu * mu;
    float inv = rsqrtf(var + 1e-5f);
    #pragma unroll
    for (int i = 0; i < 4; ++i) {
        int idx = i * 256 + t;
        o[idx] = (x[i] - mu) * inv * g[idx] + bb[idx];
    }
}

// ---------------------------------------------------------------------------
// Host orchestration
// ---------------------------------------------------------------------------
extern "C" void kernel_launch(void* const* d_in, const int* in_sizes, int n_in,
                              void* d_out, int out_size, void* d_ws, size_t ws_size,
                              hipStream_t stream) {
    (void)in_sizes; (void)n_in; (void)out_size; (void)ws_size;
    const float* q   = (const float*)d_in[0];
    const float* k   = (const float*)d_in[1];
    const float* v   = (const float*)d_in[2];
    const float* pe  = (const float*)d_in[3];
    const unsigned char* pad = (const unsigned char*)d_in[4];
    const float* qw  = (const float*)d_in[5];
    const float* kw  = (const float*)d_in[6];
    const float* kb  = (const float*)d_in[7];
    const float* vw  = (const float*)d_in[8];
    const float* vb  = (const float*)d_in[9];
    const float* rwb = (const float*)d_in[10];
    const float* rrb = (const float*)d_in[11];
    const float* rk  = (const float*)d_in[12];
    const float* pw  = (const float*)d_in[13];
    const float* pb  = (const float*)d_in[14];
    const float* lng = (const float*)d_in[15];
    const float* lnb = (const float*)d_in[16];

    const long BL = 4096, D = 1024, NHD = 1024, R2 = 2048;
    char* base = (char*)d_ws;
    size_t off = 0;
    auto alloc = [&](size_t bytes) -> char* {
        char* p = base + off;
        off = (off + bytes + 255) & ~(size_t)255;
        return p;
    };
    u16* qbf  = (u16*)alloc(BL * D * 2);       // A operands (row-major)
    u16* kbf  = (u16*)alloc(BL * D * 2);
    u16* vbf  = (u16*)alloc(BL * D * 2);
    u16* pebf = (u16*)alloc(R2 * D * 2);
    u16* qwT  = (u16*)alloc(D * NHD * 2);      // B operands, pre-transposed (N,K)
    u16* kwT  = (u16*)alloc(D * NHD * 2);
    u16* vwT  = (u16*)alloc(D * NHD * 2);
    u16* rkT  = (u16*)alloc(16L * 64 * D * 2); // (n, 64, 1024)
    u16* pwT  = (u16*)alloc(D * NHD * 2);      // (1024, 1024)
    u16* qhbf = (u16*)alloc(BL * NHD * 2);
    u16* khbf = (u16*)alloc(BL * NHD * 2);
    u16* vhbf = (u16*)alloc(BL * NHD * 2);
    u16* rbf  = (u16*)alloc(16L * R2 * 64 * 2);
    u16* obf  = (u16*)alloc(BL * NHD * 2);
    float* attf = (float*)alloc(BL * D * 4);

    auto cv = [&](const float* in, u16* outp, long nval) {
        cvt_kernel<<<1024, 256, 0, stream>>>(in, outp, nval);
    };
    auto cvT = [&](const float* in, u16* outp, long rows, long cols, long batch) {
        cvtT_kernel<<<1024, 256, 0, stream>>>(in, outp, rows, cols, batch * rows * cols);
    };
    cv(q, qbf, BL * D); cv(k, kbf, BL * D); cv(v, vbf, BL * D);
    cv(pe, pebf, R2 * D);
    cvT(qw, qwT, D, NHD, 1);        // (D,NH) -> (NH,D)
    cvT(kw, kwT, D, NHD, 1);
    cvT(vw, vwT, D, NHD, 1);
    cvT(rk, rkT, D, 64, 16);        // per head (D,64) -> (64,D)
    cvT(pw, pwT, NHD, D, 1);        // (NH,D) -> (D,NH)

    dim3 blk(256);
    // projections: (4096x1024) @ (1024x1024)
    dim3 gproj(8, 32, 1);
    gemm_bf16_kernel<false><<<gproj, blk, 0, stream>>>(qbf, 0, qwT, 0, qhbf, 0,
                                                       nullptr, BL, NHD, D, D, D, NHD);
    gemm_bf16_kernel<false><<<gproj, blk, 0, stream>>>(kbf, 0, kwT, 0, khbf, 0,
                                                       kb, BL, NHD, D, D, D, NHD);
    gemm_bf16_kernel<false><<<gproj, blk, 0, stream>>>(vbf, 0, vwT, 0, vhbf, 0,
                                                       vb, BL, NHD, D, D, D, NHD);
    // r = pos_enc @ r_kernel[n] : batched over 16 heads, (2048x1024)@(1024x64)
    dim3 grh(1, 16, 16);
    gemm_bf16_kernel<false><<<grh, blk, 0, stream>>>(pebf, 0, rkT, 64 * D, rbf,
                                                     R2 * 64, nullptr,
                                                     (int)R2, 64, (int)D, (int)D,
                                                     (int)D, 64);
    // attention: 8 query tiles x 16 heads x 4 batch
    dim3 ga(8, 16, 4);
    attn_kernel<<<ga, blk, 0, stream>>>(qhbf, khbf, vhbf, rbf, pad, rwb, rrb, obf);
    // output projection: (4096x1024)@(1024x1024) + post_b -> f32
    dim3 gpost(8, 32, 1);
    gemm_bf16_kernel<true><<<gpost, blk, 0, stream>>>(obf, 0, pwT, 0, attf, 0,
                                                      pb, BL, (int)D, (int)NHD,
                                                      (int)NHD, (int)D, (int)D);
    // residual + LayerNorm
    ln_kernel<<<4096, 256, 0, stream>>>(q, attf, lng, lnb, (float*)d_out);
}
